// MultiHeadLayer_8057358647604
// MI455X (gfx1250) — matrix operand order repaired and checked by reading, not compiled, not run on hardware
//
#include <hip/hip_runtime.h>
#include <hip/hip_bf16.h>

#define NN   6144
#define KIN  512
#define FOUT 64
#define NH   8
#define ALPHA 0.2f
#define NEG_BIG (-9.0e15f)
#define L2E 1.44269504088896340736f
#define ADJ_STRIDE 36          // padded ints per LDS adj row (conflict-free b128 reads)

typedef __attribute__((ext_vector_type(16))) __bf16 v16bf;
typedef __attribute__((ext_vector_type(8)))  float  v8f;
typedef __attribute__((ext_vector_type(4)))  float  f32x4;
typedef __attribute__((ext_vector_type(4)))  int    i32x4;
typedef __attribute__((ext_vector_type(4)))  unsigned u32x4;
typedef __attribute__((ext_vector_type(2)))  int    v2i;

struct U8 { unsigned u[8]; };

// address-space-qualified v2i pointers for the async global->LDS builtin
typedef __attribute__((address_space(1))) v2i as1_v2i;
typedef __attribute__((address_space(3))) v2i as3_v2i;

#if __has_builtin(__builtin_amdgcn_global_load_async_to_lds_b64)
#define HAVE_ASYNC 1
#else
#define HAVE_ASYNC 0
#endif

// ---- bf16 pack: v_cvt_pk_bf16_f32 when available, manual RNE fallback ------
#if __has_builtin(__builtin_amdgcn_cvt_pk_bf16_f32)
static __device__ __forceinline__ unsigned pkbf(float a, float b) {
  return __builtin_bit_cast(unsigned, __builtin_amdgcn_cvt_pk_bf16_f32(a, b));
}
static __device__ __forceinline__ unsigned bf16u(float f) {
  return __builtin_bit_cast(unsigned, __builtin_amdgcn_cvt_pk_bf16_f32(f, 0.f)) & 0xFFFFu;
}
#else
static __device__ __forceinline__ unsigned bf16u(float f) {
  unsigned u = __builtin_bit_cast(unsigned, f);
  return (u + 0x7FFFu + ((u >> 16) & 1u)) >> 16;   // round-to-nearest-even
}
static __device__ __forceinline__ unsigned pkbf(float a, float b) {
  return bf16u(a) | (bf16u(b) << 16);
}
#endif

static __device__ __forceinline__ float shflf(float v, int srcLane) {
  return __builtin_bit_cast(float,
      __builtin_amdgcn_ds_bpermute(srcLane << 2, __builtin_bit_cast(int, v)));
}
static __device__ __forceinline__ float expfast(float x) {
  return __builtin_amdgcn_exp2f(x * L2E);
}
static __device__ __forceinline__ int laneid() {
  return (int)__builtin_amdgcn_mbcnt_lo(~0u, 0u);   // wave32: lane id
}

// ---- async global->LDS copy (ASYNCcnt path), synchronous fallback ----------
static __device__ __forceinline__ void async_copy_b64(const int* g, unsigned* l) {
#if HAVE_ASYNC
  __builtin_amdgcn_global_load_async_to_lds_b64(
      (as1_v2i*)(unsigned long long)g,
      (as3_v2i*)(unsigned)(unsigned long long)l, 0, 0);
#else
  *(unsigned long long*)l = *(const unsigned long long*)g;
#endif
}
static __device__ __forceinline__ void wait_async0() {
#if HAVE_ASYNC
#if __has_builtin(__builtin_amdgcn_s_wait_asynccnt)
  __builtin_amdgcn_s_wait_asynccnt(0);
#else
  asm volatile("s_wait_asynccnt 0" ::: "memory");
#endif
#endif
}

// ---------------------------------------------------------------------------
// Kernel 1: per-head projection h = x @ W (bf16 WMMA), plus s1 = h.a1,
// s2 = h.a2, and h stored TRANSPOSED (hT[H][64][N], bf16) for kernel 2.
// grid = (N/128, NH), block = 256 (8 waves, each wave owns a 16-row tile).
// ---------------------------------------------------------------------------
__global__ __launch_bounds__(256)
void gat_proj_kernel(const float* __restrict__ x, const float* __restrict__ W,
                     const float* __restrict__ a1, const float* __restrict__ a2,
                     unsigned short* __restrict__ hT,
                     float* __restrict__ s1, float* __restrict__ s2) {
  __shared__ __align__(16) unsigned short wt[FOUT * 40];  // W chunk transposed, padded

  const int tid  = threadIdx.x;
  const int wave = tid >> 5;
  const int lane = laneid();
  const int l16  = lane & 15;
  const int hi   = lane >> 4;
  const int h    = blockIdx.y;
  const int rt   = (blockIdx.x * 8 + wave) * 16;

  v8f acc[4] = {v8f{}, v8f{}, v8f{}, v8f{}};

  const int wk  = tid >> 3;        // 0..31 : k within chunk staged by this thread
  const int wf0 = (tid & 7) * 8;   // f base staged by this thread

  const float* xr0 = x + (size_t)(rt + l16) * KIN + 8 * hi;
  const float* xr1 = xr0 + 16;

  for (int kb = 0; kb < KIN; kb += 32) {
    // --- cooperative stage of W[kb..kb+31][0..63] -> LDS (transposed, bf16)
    const float* wp = W + ((size_t)h * KIN + kb + wk) * FOUT + wf0;
    f32x4 w0 = *(const f32x4*)(wp);
    f32x4 w1 = *(const f32x4*)(wp + 4);
#pragma unroll
    for (int e = 0; e < 4; ++e) {
      wt[(wf0 + e) * 40 + wk]     = (unsigned short)bf16u(w0[e]);
      wt[(wf0 + 4 + e) * 40 + wk] = (unsigned short)bf16u(w1[e]);
    }
    __syncthreads();

    // --- A fragment from x (f32 -> bf16), layout: lane=row, k split by hi
    f32x4 r0a = *(const f32x4*)(xr0 + kb);
    f32x4 r0b = *(const f32x4*)(xr0 + kb + 4);
    f32x4 r1a = *(const f32x4*)(xr1 + kb);
    f32x4 r1b = *(const f32x4*)(xr1 + kb + 4);
    U8 af;
    af.u[0] = pkbf(r0a[0], r0a[1]); af.u[1] = pkbf(r0a[2], r0a[3]);
    af.u[2] = pkbf(r0b[0], r0b[1]); af.u[3] = pkbf(r0b[2], r0b[3]);
    af.u[4] = pkbf(r1a[0], r1a[1]); af.u[5] = pkbf(r1a[2], r1a[3]);
    af.u[6] = pkbf(r1b[0], r1b[1]); af.u[7] = pkbf(r1b[2], r1b[3]);
    v16bf av = __builtin_bit_cast(v16bf, af);

    // --- 4 B fragments from LDS, 4 WMMAs
#pragma unroll
    for (int t = 0; t < 4; ++t) {
      const u32x4* bp = (const u32x4*)&wt[(t * 16 + l16) * 40 + 8 * hi];
      u32x4 b0 = bp[0], b1 = bp[2];
      U8 bf;
      bf.u[0] = b0[0]; bf.u[1] = b0[1]; bf.u[2] = b0[2]; bf.u[3] = b0[3];
      bf.u[4] = b1[0]; bf.u[5] = b1[1]; bf.u[6] = b1[2]; bf.u[7] = b1[3];
      v16bf bv = __builtin_bit_cast(v16bf, bf);
      acc[t] = __builtin_amdgcn_wmma_f32_16x16x32_bf16(
          false, av, false, bv, (short)0, acc[t], false, false);
    }
    __syncthreads();
  }

  // --- s1 / s2: per-row dot with a1, a2 (C layout: vgpr r -> row r+8*hi, lane l16 -> col)
  float a1v[4], a2v[4];
#pragma unroll
  for (int t = 0; t < 4; ++t) {
    a1v[t] = a1[h * FOUT + t * 16 + l16];
    a2v[t] = a2[h * FOUT + t * 16 + l16];
  }
  float p1r[8], p2r[8];
#pragma unroll
  for (int r = 0; r < 8; ++r) { p1r[r] = 0.f; p2r[r] = 0.f; }
#pragma unroll
  for (int t = 0; t < 4; ++t)
#pragma unroll
    for (int r = 0; r < 8; ++r) {
      p1r[r] += acc[t][r] * a1v[t];
      p2r[r] += acc[t][r] * a2v[t];
    }
#pragma unroll
  for (int msk = 1; msk <= 8; msk <<= 1)
#pragma unroll
    for (int r = 0; r < 8; ++r) {
      p1r[r] += shflf(p1r[r], lane ^ msk);
      p2r[r] += shflf(p2r[r], lane ^ msk);
    }
  if (l16 == 0) {
#pragma unroll
    for (int r = 0; r < 8; ++r) {
      s1[h * NN + rt + 8 * hi + r] = p1r[r];
      s2[h * NN + rt + 8 * hi + r] = p2r[r];
    }
  }

  // --- store h transposed as bf16: hT[(h*64+f)*N + node]
#pragma unroll
  for (int t = 0; t < 4; ++t) {
    u32x4 pk;
    pk[0] = pkbf(acc[t][0], acc[t][1]);
    pk[1] = pkbf(acc[t][2], acc[t][3]);
    pk[2] = pkbf(acc[t][4], acc[t][5]);
    pk[3] = pkbf(acc[t][6], acc[t][7]);
    *(u32x4*)&hT[(size_t)(h * FOUT + t * 16 + l16) * NN + rt + 8 * hi] = pk;
  }
}

// ---------------------------------------------------------------------------
// Kernel 2: fused masked-softmax attention + att@h (flash-style online
// softmax) + normalization + ELU + concat-heads store.
// grid = N/16, block = 256 (wave w == head w). The 16x32 adj tile is staged
// into LDS ONCE per block via async global->LDS copies (double-buffered, DMA
// overlapped with softmax+WMMA of the current tile), so adj is read from HBM
// exactly once.
// ---------------------------------------------------------------------------
__global__ __launch_bounds__(256)
void gat_attn_kernel(const int* __restrict__ adj,
                     const unsigned short* __restrict__ hT,
                     const float* __restrict__ s1, const float* __restrict__ s2,
                     float* __restrict__ out) {
  __shared__ __align__(16) unsigned adjbuf[2 * 16 * ADJ_STRIDE];

  const int tid  = threadIdx.x;
  const int h    = tid >> 5;
  const int lane = laneid();
  const int l16  = lane & 15;
  const int hi   = lane >> 4;
  const int rb   = blockIdx.x;
  const int row  = rb * 16 + l16;

  const float s1v = s1[h * NN + row];
  const float* s2p = s2 + h * NN + 8 * hi;
  const unsigned short* hTh = hT + (size_t)h * FOUT * NN;

  // staging assignment: thread copies 2 ints (b64) of the 16x32 tile
  const int srow = tid >> 4;           // 0..15
  const int scol = (tid & 15) * 2;     // 0,2,..,30
  const int* srcBase = adj + (size_t)(rb * 16 + srow) * NN + scol;

  v8f acc[4] = {v8f{}, v8f{}, v8f{}, v8f{}};
  float m = NEG_BIG, lsum = 0.f;

  // preload chunk 0
  async_copy_b64(srcBase, &adjbuf[srow * ADJ_STRIDE + scol]);
  wait_async0();
  __syncthreads();

  for (int cb = 0, ci = 0; cb < NN; cb += 32, ci ^= 1) {
    // kick off DMA of the next adj tile into the other buffer
    if (cb + 32 < NN)
      async_copy_b64(srcBase + cb + 32,
                     &adjbuf[(ci ^ 1) * (16 * ADJ_STRIDE) + srow * ADJ_STRIDE + scol]);

    // scores for 16 k-values owned by this lane (A-fragment layout):
    // run0: k = 8*hi + j, run1: k = 16 + 8*hi + j
    f32x4 f0 = *(const f32x4*)(s2p + cb);
    f32x4 f1 = *(const f32x4*)(s2p + cb + 4);
    f32x4 g0 = *(const f32x4*)(s2p + cb + 16);
    f32x4 g1 = *(const f32x4*)(s2p + cb + 20);
    const unsigned* abuf = &adjbuf[ci * (16 * ADJ_STRIDE) + l16 * ADJ_STRIDE + 8 * hi];
    i32x4 a0  = *(const i32x4*)(abuf);
    i32x4 a1i = *(const i32x4*)(abuf + 4);
    i32x4 b0  = *(const i32x4*)(abuf + 16);
    i32x4 b1i = *(const i32x4*)(abuf + 20);

    float p0[8], p1[8];
    float cm = NEG_BIG;
#pragma unroll
    for (int j = 0; j < 4; ++j) {
      float t0 = s1v + f0[j];
      t0 = fmaxf(t0, 0.f) + ALPHA * fminf(t0, 0.f);
      t0 = (a0[j] > 0) ? t0 : NEG_BIG;
      p0[j] = t0; cm = fmaxf(cm, t0);
      float t1 = s1v + f1[j];
      t1 = fmaxf(t1, 0.f) + ALPHA * fminf(t1, 0.f);
      t1 = (a1i[j] > 0) ? t1 : NEG_BIG;
      p0[4 + j] = t1; cm = fmaxf(cm, t1);
      float t2 = s1v + g0[j];
      t2 = fmaxf(t2, 0.f) + ALPHA * fminf(t2, 0.f);
      t2 = (b0[j] > 0) ? t2 : NEG_BIG;
      p1[j] = t2; cm = fmaxf(cm, t2);
      float t3 = s1v + g1[j];
      t3 = fmaxf(t3, 0.f) + ALPHA * fminf(t3, 0.f);
      t3 = (b1i[j] > 0) ? t3 : NEG_BIG;
      p1[4 + j] = t3; cm = fmaxf(cm, t3);
    }
    // combine row max across the two half-row lanes
    cm = fmaxf(cm, shflf(cm, lane ^ 16));
    float mnew  = fmaxf(m, cm);
    float scale = expfast(m - mnew);
    float psum = 0.f;
#pragma unroll
    for (int j = 0; j < 8; ++j) {
      p0[j] = expfast(p0[j] - mnew); psum += p0[j];
      p1[j] = expfast(p1[j] - mnew); psum += p1[j];
    }
    psum += shflf(psum, lane ^ 16);
    lsum = lsum * scale + psum;
    m = mnew;

    // pack probabilities into bf16 A fragment
    U8 af;
    af.u[0] = pkbf(p0[0], p0[1]); af.u[1] = pkbf(p0[2], p0[3]);
    af.u[2] = pkbf(p0[4], p0[5]); af.u[3] = pkbf(p0[6], p0[7]);
    af.u[4] = pkbf(p1[0], p1[1]); af.u[5] = pkbf(p1[2], p1[3]);
    af.u[6] = pkbf(p1[4], p1[5]); af.u[7] = pkbf(p1[6], p1[7]);
    v16bf av = __builtin_bit_cast(v16bf, af);

    // rescale accumulator rows only when the running max moved (wave-uniform)
#if __has_builtin(__builtin_amdgcn_ballot_w32)
    bool needScale = __builtin_amdgcn_ballot_w32(scale != 1.0f) != 0u;
#else
    bool needScale = true;
#endif
    if (needScale) {
      float scr[8];
#pragma unroll
      for (int r = 0; r < 8; ++r) scr[r] = shflf(scale, r + 8 * hi);
#pragma unroll
      for (int t = 0; t < 4; ++t)
#pragma unroll
        for (int r = 0; r < 8; ++r) acc[t][r] *= scr[r];
    }

    // B fragments: table-order bf16 from hT (k = 8*hi + j, 16 + 8*hi + j), then 4 WMMAs
#pragma unroll
    for (int t = 0; t < 4; ++t) {
      const u32x4* bp =
          (const u32x4*)(hTh + (size_t)(t * 16 + l16) * NN + cb + 8 * hi);
      u32x4 b0v = bp[0], b1v = bp[2];
      U8 bf;
      bf.u[0] = b0v[0]; bf.u[1] = b0v[1]; bf.u[2] = b0v[2]; bf.u[3] = b0v[3];
      bf.u[4] = b1v[0]; bf.u[5] = b1v[1]; bf.u[6] = b1v[2]; bf.u[7] = b1v[3];
      v16bf bv = __builtin_bit_cast(v16bf, bf);
      acc[t] = __builtin_amdgcn_wmma_f32_16x16x32_bf16(
          false, av, false, bv, (short)0, acc[t], false, false);
    }

    // next tile DMA done + everyone finished reading current tile
    wait_async0();
    __syncthreads();
  }

  // epilogue: normalize, ELU, store concat [N, H*F]
  float rinv = 1.0f / lsum;
#pragma unroll
  for (int r = 0; r < 8; ++r) {
    float iv = shflf(rinv, r + 8 * hi);
    const int node = rb * 16 + r + 8 * hi;
#pragma unroll
    for (int t = 0; t < 4; ++t) {
      float o = acc[t][r] * iv;
      o = (o > 0.f) ? o : (expfast(o) - 1.f);
      out[(size_t)node * (NH * FOUT) + h * FOUT + t * 16 + l16] = o;
    }
  }
}

// ---------------------------------------------------------------------------
extern "C" void kernel_launch(void* const* d_in, const int* in_sizes, int n_in,
                              void* d_out, int out_size, void* d_ws, size_t ws_size,
                              hipStream_t stream) {
  const float* x   = (const float*)d_in[0];
  const int*   adj = (const int*)d_in[1];
  const float* W   = (const float*)d_in[2];
  const float* a1  = (const float*)d_in[3];
  const float* a2  = (const float*)d_in[4];
  float* out = (float*)d_out;

  unsigned short* hT = (unsigned short*)d_ws;                       // [H][64][N] bf16
  float* s1 = (float*)((char*)d_ws + (size_t)NH * FOUT * NN * 2);   // [H][N] f32
  float* s2 = s1 + (size_t)NH * NN;                                 // [H][N] f32

  gat_proj_kernel<<<dim3(NN / 128, NH), 256, 0, stream>>>(x, W, a1, a2, hT, s1, s2);
  gat_attn_kernel<<<dim3(NN / 16), 256, 0, stream>>>(adj, hT, s1, s2, out);
}
